// AdjModel_9766755631707
// MI455X (gfx1250) — matrix operand order repaired and checked by reading, not compiled, not run on hardware
//
#include <hip/hip_runtime.h>
#include <hip/hip_bf16.h>
#include <math.h>

// ---------------------------------------------------------------------------
// Problem constants (match reference)
// ---------------------------------------------------------------------------
#define NN   8192
#define DD   16
#define BB   4
#define FENC 128
#define NH   4
#define BN   (BB * NN)            // 32768 rows total
#define FLOW_ITERS 20
#define DUAL_ITERS 20
#define DUAL_STEP 0.01f
#define DUAL_MOM  0.9f

typedef _Float16 v16h __attribute__((ext_vector_type(16)));
typedef float    v8f  __attribute__((ext_vector_type(8)));

// ---------------------------------------------------------------------------
// Fast transcendentals: gfx1250 has v_tanh_f32 (TRANS unit, co-executes with
// WMMA).  Guarded so the code still compiles if the builtin is absent.
// ---------------------------------------------------------------------------
__device__ __forceinline__ float fast_tanhf(float x) {
#if __has_builtin(__builtin_amdgcn_tanhf)
  return __builtin_amdgcn_tanhf(x);
#else
  float t = __expf(-2.0f * fabsf(x));
  float r = (1.0f - t) / (1.0f + t);
  return copysignf(r, x);
#endif
}
__device__ __forceinline__ float sigmoidf_(float x) {
  return 1.0f / (1.0f + __expf(-x));
}

// ---------------------------------------------------------------------------
// Async global->LDS DMA (GLOBAL_LOAD_ASYNC_TO_LDS_B128, ASYNCcnt-tracked).
// Builtin signature (from hipcc diagnostic): pointers to int vector_size(16),
// global operand in AS(1), LDS operand in AS(3).
// Guarded: falls back to plain float4 copies if the builtin is unavailable.
// ---------------------------------------------------------------------------
#if defined(__AMDGCN__) && __has_builtin(__builtin_amdgcn_global_load_async_to_lds_b128)
#define USE_ASYNC_LDS 1
#else
#define USE_ASYNC_LDS 0
#endif

typedef int vi4 __attribute__((vector_size(16)));
typedef __attribute__((address_space(1))) vi4* gvi4p;
typedef __attribute__((address_space(3))) vi4* lvi4p;

__device__ __forceinline__ void async_copy16(const float* g, float* l) {
#if USE_ASYNC_LDS
  __builtin_amdgcn_global_load_async_to_lds_b128((gvi4p)(void*)g, (lvi4p)(void*)l,
                                                 0, 0);
#else
  ((float4*)l)[0] = ((const float4*)g)[0];
#endif
}
__device__ __forceinline__ void async_wait0() {
#if USE_ASYNC_LDS
#if __has_builtin(__builtin_amdgcn_s_wait_asynccnt)
  __builtin_amdgcn_s_wait_asynccnt(0);
#else
  asm volatile("s_wait_asynccnt 0x0" ::: "memory");
#endif
#endif
}

// ---------------------------------------------------------------------------
// WMMA helpers: CDNA5 v_wmma_f32_16x16x32_f16, wave32 fragment layouts
//   A (16x32 f16):  lane: row = lane&15, half = lane>>4
//                   elem i -> k = k0 + ((i&8)<<1) + half*8 + (i&7)
//   B (32x16 f16):  lane: col = lane&15; elem i -> k = k0 + ((i&8)<<1) + (lane>>4)*8 + (i&7)
//   C/D (16x16 f32): vgpr r, lane: row = r + (lane>>4)*8, col = lane&15
// ---------------------------------------------------------------------------
__device__ __forceinline__ v8f wmma16(v16h a, v16h b, v8f c) {
  return __builtin_amdgcn_wmma_f32_16x16x32_f16(false, a, false, b,
                                                (short)0, c, false, false);
}

__device__ __forceinline__ v16h afrag_lds(const _Float16* A, int ld, int k0, int lane) {
  int row = lane & 15, half = lane >> 4;
  v16h a;
#pragma unroll
  for (int i = 0; i < 16; ++i) {
    int k = k0 + ((i & 8) << 1) + half * 8 + (i & 7);
    a[i] = A[row * ld + k];
  }
  return a;
}

__device__ __forceinline__ v16h afrag_f32(const float* A, int ld, int k0, int lane) {
  int row = lane & 15, half = lane >> 4;
  v16h a;
#pragma unroll
  for (int i = 0; i < 16; ++i) {
    int k = k0 + ((i & 8) << 1) + half * 8 + (i & 7);
    a[i] = (_Float16)A[(size_t)row * ld + k];
  }
  return a;
}

// B fragment from global f32 row-major weight [K x ld], cols n0..n0+15,
// rows k0..k0+31 (rows >= kmax are zero-padded).
__device__ __forceinline__ v16h bfrag_g(const float* W, int ld, int k0, int n0,
                                        int kmax, int lane) {
  int n  = n0 + (lane & 15);
  int kb = k0 + ((lane >> 4) << 3);
  v16h b;
#pragma unroll
  for (int i = 0; i < 16; ++i) {
    int k = kb + ((i & 8) << 1) + (i & 7);
    b[i] = (k < kmax) ? (_Float16)W[(size_t)k * ld + n] : (_Float16)0.0f;
  }
  return b;
}

// ---------------------------------------------------------------------------
// 0) zero the accumulators (12 floats)
// ---------------------------------------------------------------------------
__global__ void k_zero_acc(float* acc) {
  if (threadIdx.x < 12) acc[threadIdx.x] = 0.0f;
}

// ---------------------------------------------------------------------------
// 1) Encoder: h = tanh(tanh([emb|feat] @ W1 + b1) @ W2 + b2)   -> hS [BN,128]
// ---------------------------------------------------------------------------
__device__ __forceinline__ v16h enc_afrag(const float* emb, const float* feat,
                                          size_t m0, int k0, int lane) {
  size_t row = m0 + (lane & 15);
  int half = lane >> 4;
  v16h a;
#pragma unroll
  for (int i = 0; i < 16; ++i) {
    int k = k0 + ((i & 8) << 1) + half * 8 + (i & 7);
    float v = 0.0f;
    if (k < 32)      v = emb[row * 32 + k];
    else if (k < 36) v = feat[row * 4 + (k - 32)];
    a[i] = (_Float16)v;
  }
  return a;
}

__global__ void __launch_bounds__(128)
k_encoder(const float* __restrict__ emb, const float* __restrict__ feat,
          const float* __restrict__ W1, const float* __restrict__ b1,
          const float* __restrict__ W2, const float* __restrict__ b2,
          float* __restrict__ hS) {
  __shared__ _Float16 h1s[4][16 * 64];
  const int lane = threadIdx.x & 31, wave = threadIdx.x >> 5;
  const int tile = blockIdx.x * 4 + wave;
  const size_t m0 = (size_t)tile * 16;
  const int nidx = lane & 15, half = lane >> 4;
  _Float16* h1 = h1s[wave];

  v16h a0 = enc_afrag(emb, feat, m0, 0, lane);
  v16h a1 = enc_afrag(emb, feat, m0, 32, lane);
  for (int nt = 0; nt < 4; ++nt) {
    float bias = b1[nt * 16 + nidx];
    v8f c;
#pragma unroll
    for (int r = 0; r < 8; ++r) c[r] = bias;
    c = wmma16(a0, bfrag_g(W1, 64, 0, nt * 16, 36, lane), c);
    c = wmma16(a1, bfrag_g(W1, 64, 32, nt * 16, 36, lane), c);
#pragma unroll
    for (int r = 0; r < 8; ++r)
      h1[(r + half * 8) * 64 + nt * 16 + nidx] = (_Float16)fast_tanhf(c[r]);
  }
  v16h a2 = afrag_lds(h1, 64, 0, lane);
  v16h a3 = afrag_lds(h1, 64, 32, lane);
  for (int nt = 0; nt < 8; ++nt) {
    float bias = b2[nt * 16 + nidx];
    v8f c;
#pragma unroll
    for (int r = 0; r < 8; ++r) c[r] = bias;
    c = wmma16(a2, bfrag_g(W2, 128, 0, nt * 16, 64, lane), c);
    c = wmma16(a3, bfrag_g(W2, 128, 32, nt * 16, 64, lane), c);
#pragma unroll
    for (int r = 0; r < 8; ++r)
      hS[(m0 + r + half * 8) * 128 + nt * 16 + nidx] = fast_tanhf(c[r]);
  }
}

// ---------------------------------------------------------------------------
// 2) wq[h,f] = sum_g W[h,f,g]*aq[h,g]; wk likewise.  512 entries.
// ---------------------------------------------------------------------------
__global__ void k_wqk(const float* __restrict__ W, const float* __restrict__ aq,
                      const float* __restrict__ ak,
                      float* __restrict__ wq, float* __restrict__ wk) {
  int tid = blockIdx.x * blockDim.x + threadIdx.x;
  if (tid >= NH * FENC) return;
  int h = tid >> 7, f = tid & 127;
  const float* wr = W + ((size_t)h * FENC + f) * FENC;
  const float* q = aq + h * FENC;
  const float* k = ak + h * FENC;
  float sq = 0.f, sk = 0.f;
  for (int g = 0; g < FENC; ++g) { float w = wr[g]; sq += w * q[g]; sk += w * k[g]; }
  wq[tid] = sq; wk[tid] = sk;
}

// ---------------------------------------------------------------------------
// 3) cq[b,h,n] = h[b,n,:] . wq[h];  ck likewise.
// ---------------------------------------------------------------------------
__global__ void k_cqck(const float* __restrict__ hS, const float* __restrict__ wq,
                       const float* __restrict__ wk,
                       float* __restrict__ cq, float* __restrict__ ck) {
  size_t tid = (size_t)blockIdx.x * blockDim.x + threadIdx.x;  // (b*4+h)*N + n
  int n = (int)(tid & (NN - 1));
  int bh = (int)(tid >> 13);
  int h = bh & 3, b = bh >> 2;
  const float* row = hS + ((size_t)b * NN + n) * 128;
  const float* q = wq + h * 128;
  const float* k = wk + h * 128;
  float sq = 0.f, sk = 0.f;
  for (int f = 0; f < 128; ++f) { float v = row[f]; sq += v * q[f]; sk += v * k[f]; }
  cq[tid] = sq; ck[tid] = sk;
}

// ---------------------------------------------------------------------------
// 4) Attention: neighbor softmax + weighted sum (->LDS f16), then WMMA GEMM
//    out = tanh( mean_h( weighted_h @ W[h] ) ).   2 waves / block, 32KB LDS.
// ---------------------------------------------------------------------------
__global__ void __launch_bounds__(64)
k_attn(const float* __restrict__ hS, const int* __restrict__ adj,
       const float* __restrict__ cq, const float* __restrict__ ck,
       const float* __restrict__ attnW, float* __restrict__ nxt) {
  __shared__ _Float16 wsm[2][NH * 16 * 128];   // weighted, f16, per wave
  const int lane = threadIdx.x & 31, wave = threadIdx.x >> 5;
  const int tile = blockIdx.x * 2 + wave;
  const size_t m0 = (size_t)tile * 16;
  const int b = (int)(m0 >> 13);
  const int n0 = (int)(m0 & (NN - 1));
  const int node = lane & 15, fh = lane >> 4;
  _Float16* wl = wsm[wave];

  int nb[DD];
  const int* arow = adj + ((size_t)b * NN + (n0 + node)) * DD;
#pragma unroll
  for (int d = 0; d < DD; ++d) nb[d] = arow[d];

  for (int h = 0; h < NH; ++h) {
    const float* ckh = ck + (size_t)(b * NH + h) * NN;
    float cqv = cq[(size_t)(b * NH + h) * NN + n0 + node];
    float e[DD], mx = -1e30f;
#pragma unroll
    for (int d = 0; d < DD; ++d) {
      e[d] = fast_tanhf(cqv + ckh[nb[d]]);
      mx = fmaxf(mx, e[d]);
    }
    float s = 0.f;
#pragma unroll
    for (int d = 0; d < DD; ++d) { e[d] = __expf(e[d] - mx); s += e[d]; }
    float inv = 1.0f / s;

    float4 acc[16];
#pragma unroll
    for (int q = 0; q < 16; ++q) acc[q] = make_float4(0.f, 0.f, 0.f, 0.f);
    for (int d = 0; d < DD; ++d) {
      float w = e[d] * inv;
      const float4* src =
          (const float4*)(hS + ((size_t)b * NN + nb[d]) * 128 + fh * 64);
#pragma unroll
      for (int q = 0; q < 16; ++q) {
        float4 v = src[q];
        acc[q].x += w * v.x; acc[q].y += w * v.y;
        acc[q].z += w * v.z; acc[q].w += w * v.w;
      }
    }
    _Float16* dst = wl + h * (16 * 128) + node * 128 + fh * 64;
#pragma unroll
    for (int q = 0; q < 16; ++q) {
      dst[4 * q + 0] = (_Float16)acc[q].x; dst[4 * q + 1] = (_Float16)acc[q].y;
      dst[4 * q + 2] = (_Float16)acc[q].z; dst[4 * q + 3] = (_Float16)acc[q].w;
    }
  }

  for (int nt = 0; nt < 8; ++nt) {
    v8f c;
#pragma unroll
    for (int r = 0; r < 8; ++r) c[r] = 0.f;
    for (int h = 0; h < NH; ++h) {
      const float* Wh = attnW + (size_t)h * FENC * FENC;
      for (int kc = 0; kc < 4; ++kc) {
        v16h a = afrag_lds(wl + h * (16 * 128), 128, kc * 32, lane);
        v16h bb = bfrag_g(Wh, 128, kc * 32, nt * 16, 128, lane);
        c = wmma16(a, bb, c);
      }
    }
#pragma unroll
    for (int r = 0; r < 8; ++r)
      nxt[(m0 + r + fh * 8) * 128 + nt * 16 + node] = fast_tanhf(c[r] * 0.25f);
  }
}

// ---------------------------------------------------------------------------
// 5) GRU: h <- z*h + (1-z)*tanh(gc + (r*h)@Uc), gates via WMMA.
//    x/h tiles staged f32 via async global->LDS DMA.  2 waves/block, 40KB LDS.
// ---------------------------------------------------------------------------
__global__ void __launch_bounds__(64)
k_gru(const float* __restrict__ nxt, float* __restrict__ hS,
      const float* __restrict__ W, const float* __restrict__ U,
      const float* __restrict__ bv) {
  __shared__ __align__(16) float xls[2][16 * 128];
  __shared__ __align__(16) float hls[2][16 * 128];
  __shared__ _Float16 rls[2][16 * 128];
  const int lane = threadIdx.x & 31, wave = threadIdx.x >> 5;
  const int tile = blockIdx.x * 2 + wave;
  const size_t m0 = (size_t)tile * 16;
  const int nidx = lane & 15, half = lane >> 4;
  float* xl = xls[wave];
  float* hl = hls[wave];
  _Float16* rl = rls[wave];

  // async DMA: 512 16-byte chunks per tile for each of x and h
  for (int t = lane; t < 512; t += 32) {
    async_copy16(nxt + m0 * 128 + t * 4, xl + t * 4);
    async_copy16(hS + m0 * 128 + t * 4, hl + t * 4);
  }
  async_wait0();

  v16h ax[4], ah[4];
#pragma unroll
  for (int kc = 0; kc < 4; ++kc) {
    ax[kc] = afrag_f32(xl, 128, kc * 32, lane);
    ah[kc] = afrag_f32(hl, 128, kc * 32, lane);
  }

  // phase 1: r gate, write r*h to LDS
  for (int ft = 0; ft < 8; ++ft) {
    float bias = bv[128 + ft * 16 + nidx];
    v8f c;
#pragma unroll
    for (int r = 0; r < 8; ++r) c[r] = bias;
    for (int kc = 0; kc < 4; ++kc)
      c = wmma16(ax[kc], bfrag_g(W, 384, kc * 32, 128 + ft * 16, 128, lane), c);
    for (int kc = 0; kc < 4; ++kc)
      c = wmma16(ah[kc], bfrag_g(U, 384, kc * 32, 128 + ft * 16, 128, lane), c);
#pragma unroll
    for (int r = 0; r < 8; ++r) {
      int rr = r + half * 8;
      float rv = sigmoidf_(c[r]);
      float hv = hl[rr * 128 + ft * 16 + nidx];
      rl[rr * 128 + ft * 16 + nidx] = (_Float16)(rv * hv);
    }
  }

  v16h ar[4];
#pragma unroll
  for (int kc = 0; kc < 4; ++kc) ar[kc] = afrag_lds(rl, 128, kc * 32, lane);

  // phase 2: z and c gates, combine
  for (int ft = 0; ft < 8; ++ft) {
    float bz = bv[ft * 16 + nidx];
    float bc = bv[256 + ft * 16 + nidx];
    v8f cz, cc;
#pragma unroll
    for (int r = 0; r < 8; ++r) { cz[r] = bz; cc[r] = bc; }
    for (int kc = 0; kc < 4; ++kc)
      cz = wmma16(ax[kc], bfrag_g(W, 384, kc * 32, ft * 16, 128, lane), cz);
    for (int kc = 0; kc < 4; ++kc)
      cz = wmma16(ah[kc], bfrag_g(U, 384, kc * 32, ft * 16, 128, lane), cz);
    for (int kc = 0; kc < 4; ++kc)
      cc = wmma16(ax[kc], bfrag_g(W, 384, kc * 32, 256 + ft * 16, 128, lane), cc);
    for (int kc = 0; kc < 4; ++kc)
      cc = wmma16(ar[kc], bfrag_g(U, 384, kc * 32, 256 + ft * 16, 128, lane), cc);
#pragma unroll
    for (int r = 0; r < 8; ++r) {
      int rr = r + half * 8;
      float z = sigmoidf_(cz[r]);
      float cv = fast_tanhf(cc[r]);
      float hv = hl[rr * 128 + ft * 16 + nidx];
      hS[(m0 + rr) * 128 + ft * 16 + nidx] = z * hv + (1.0f - z) * cv;
    }
  }
}

// ---------------------------------------------------------------------------
// 6) Head MLP (decoder / dual): out[m] = tanh(h@W1+b1) @ W2 + b2
// ---------------------------------------------------------------------------
__global__ void __launch_bounds__(128)
k_headmlp(const float* __restrict__ hS, const float* __restrict__ W1,
          const float* __restrict__ b1, const float* __restrict__ W2,
          const float* __restrict__ b2, float* __restrict__ outv) {
  __shared__ float hid[4][16 * 64];
  const int lane = threadIdx.x & 31, wave = threadIdx.x >> 5;
  const int tile = blockIdx.x * 4 + wave;
  const size_t m0 = (size_t)tile * 16;
  const int nidx = lane & 15, half = lane >> 4;
  float* hd = hid[wave];

  v16h a[4];
#pragma unroll
  for (int kc = 0; kc < 4; ++kc)
    a[kc] = afrag_f32(hS + m0 * 128, 128, kc * 32, lane);

  for (int nt = 0; nt < 4; ++nt) {
    float bias = b1[nt * 16 + nidx];
    v8f c;
#pragma unroll
    for (int r = 0; r < 8; ++r) c[r] = bias;
    for (int kc = 0; kc < 4; ++kc)
      c = wmma16(a[kc], bfrag_g(W1, 64, kc * 32, nt * 16, 128, lane), c);
#pragma unroll
    for (int r = 0; r < 8; ++r)
      hd[(r + half * 8) * 64 + nt * 16 + nidx] = fast_tanhf(c[r]);
  }
  if (lane < 16) {
    float s = b2[0];
    for (int f = 0; f < 64; ++f) s += hd[lane * 64 + f] * W2[f];
    outv[m0 + lane] = s;
  }
}

// ---------------------------------------------------------------------------
// 7) pred = node_w[adj]; norm_w = softmax_d(pred); flow0 = norm_w*relu(-dem)
// ---------------------------------------------------------------------------
__global__ void k_pred_softmax(const float* __restrict__ node_w,
                               const int* __restrict__ adj,
                               const float* __restrict__ dem,
                               float* __restrict__ norm_w,
                               float* __restrict__ flow0) {
  size_t tid = (size_t)blockIdx.x * blockDim.x + threadIdx.x;  // b*N + n
  int b = (int)(tid >> 13);
  const int* arow = adj + tid * DD;
  float p[DD], mx = -1e30f;
#pragma unroll
  for (int d = 0; d < DD; ++d) {
    p[d] = node_w[(size_t)b * NN + arow[d]];
    mx = fmaxf(mx, p[d]);
  }
  float s = 0.f;
#pragma unroll
  for (int d = 0; d < DD; ++d) { p[d] = __expf(p[d] - mx); s += p[d]; }
  float inv = 1.0f / s;
  float f0 = fmaxf(-dem[tid], 0.0f);
#pragma unroll
  for (int d = 0; d < DD; ++d) {
    float w = p[d] * inv;
    norm_w[tid * DD + d] = w;
    flow0[tid * DD + d] = w * f0;
  }
}

// ---------------------------------------------------------------------------
// 8) one MCF iteration: inflow[v] = sum_d flow[src(v,d), d];
//    new = norm_w * relu(inflow - demand)
// ---------------------------------------------------------------------------
__global__ void k_mcf(const float* __restrict__ src, const int* __restrict__ inidx,
                      const float* __restrict__ norm_w, const float* __restrict__ dem,
                      float* __restrict__ dst) {
  size_t tid = (size_t)blockIdx.x * blockDim.x + threadIdx.x;  // b*N + n
  int b = (int)(tid >> 13);
  const int* irow = inidx + tid * DD;
  __builtin_prefetch(norm_w + tid * DD, 0, 1);
  float s = 0.f;
#pragma unroll
  for (int d = 0; d < DD; ++d)
    s += src[((size_t)b * NN + irow[d]) * DD + d];
  float x = fmaxf(s - dem[tid], 0.0f);
#pragma unroll
  for (int d = 0; d < DD; ++d)
    dst[tid * DD + d] = norm_w[tid * DD + d] * x;
}

// ---------------------------------------------------------------------------
// 9) reverse-edge correction + sum of cost(flow) -> acc[b]
// ---------------------------------------------------------------------------
__global__ void k_flowcost(const float* __restrict__ flow,
                           const int* __restrict__ outidx,
                           float* __restrict__ acc) {
  __shared__ float red[256];
  size_t tid = (size_t)blockIdx.x * blockDim.x + threadIdx.x;  // over B*N*D
  int b = (int)(tid >> 17);                                    // N*D = 2^17
  float f = flow[tid];
  float rev = flow[((size_t)b << 17) + outidx[tid]];
  float fc = fmaxf(f - fminf(f, rev), 0.0f);
  red[threadIdx.x] = fc * fc;
  __syncthreads();
  for (int s = 128; s > 0; s >>= 1) {
    if ((int)threadIdx.x < s) red[threadIdx.x] += red[threadIdx.x + s];
    __syncthreads();
  }
  if (threadIdx.x == 0) atomicAdd(&acc[b], red[0]);
}

// ---------------------------------------------------------------------------
// 10) dual_diff + 20-iter projected momentum descent + dual cost -> acc[4+b]
// ---------------------------------------------------------------------------
__global__ void k_dual(const float* __restrict__ dualv, const int* __restrict__ adj,
                       float* __restrict__ acc) {
  __shared__ float red[256];
  size_t tid = (size_t)blockIdx.x * blockDim.x + threadIdx.x;  // over B*N*D
  int b = (int)(tid >> 17);
  int n = (int)((tid >> 4) & (NN - 1));
  int a = adj[tid];
  float mask = (a != NN) ? 1.0f : 0.0f;
  float dd = mask * (dualv[(size_t)b * NN + n] - dualv[(size_t)b * NN + a]);
  float fl = 0.f, ac = 0.f;
  for (int it = 0; it < DUAL_ITERS; ++it) {
    float deriv = 2.0f * (fl - DUAL_MOM * ac) - dd;
    ac = DUAL_MOM * ac + DUAL_STEP * deriv;
    fl = fmaxf(mask * (fl - ac), 0.0f);
  }
  red[threadIdx.x] = fl * fl - dd * fl;
  __syncthreads();
  for (int s = 128; s > 0; s >>= 1) {
    if ((int)threadIdx.x < s) red[threadIdx.x] += red[threadIdx.x + s];
    __syncthreads();
  }
  if (threadIdx.x == 0) atomicAdd(&acc[4 + b], red[0]);
}

// ---------------------------------------------------------------------------
// 11) dual_demand = sum_n dual_vars * demands -> acc[8+b]
// ---------------------------------------------------------------------------
__global__ void k_dualdemand(const float* __restrict__ dualv,
                             const float* __restrict__ dem,
                             float* __restrict__ acc) {
  __shared__ float red[256];
  size_t tid = (size_t)blockIdx.x * blockDim.x + threadIdx.x;  // b*N + n
  int b = (int)(tid >> 13);
  red[threadIdx.x] = dualv[tid] * dem[tid];
  __syncthreads();
  for (int s = 128; s > 0; s >>= 1) {
    if ((int)threadIdx.x < s) red[threadIdx.x] += red[threadIdx.x + s];
    __syncthreads();
  }
  if (threadIdx.x == 0) atomicAdd(&acc[8 + b], red[0]);
}

// ---------------------------------------------------------------------------
// 12) out[b] = flow_cost - (dual_flow_cost - dual_demand)
// ---------------------------------------------------------------------------
__global__ void k_final(const float* __restrict__ acc, float* __restrict__ out) {
  int b = threadIdx.x;
  if (b < BB) out[b] = acc[b] - acc[4 + b] + acc[8 + b];
}

// ---------------------------------------------------------------------------
// Host launcher
// ---------------------------------------------------------------------------
extern "C" void kernel_launch(void* const* d_in, const int* in_sizes, int n_in,
                              void* d_out, int out_size, void* d_ws, size_t ws_size,
                              hipStream_t stream) {
  const float* emb   = (const float*)d_in[0];
  const float* feat  = (const float*)d_in[1];
  const float* dem   = (const float*)d_in[2];
  const int*   adj   = (const int*)d_in[3];
  const int*   inidx = (const int*)d_in[4];
  const int*   outix = (const int*)d_in[5];
  const float* encW1 = (const float*)d_in[7];
  const float* encb1 = (const float*)d_in[8];
  const float* encW2 = (const float*)d_in[9];
  const float* encb2 = (const float*)d_in[10];
  const float* attnW = (const float*)d_in[11];
  const float* aq    = (const float*)d_in[12];
  const float* ak    = (const float*)d_in[13];
  const float* gruW  = (const float*)d_in[14];
  const float* gruU  = (const float*)d_in[15];
  const float* grub  = (const float*)d_in[16];
  const float* decW1 = (const float*)d_in[17];
  const float* decb1 = (const float*)d_in[18];
  const float* decW2 = (const float*)d_in[19];
  const float* decb2 = (const float*)d_in[20];
  const float* dW1   = (const float*)d_in[21];
  const float* db1   = (const float*)d_in[22];
  const float* dW2   = (const float*)d_in[23];
  const float* db2   = (const float*)d_in[24];

  float* ws = (float*)d_ws;
  float* hS     = ws;                       // BN*128
  float* nxt    = hS + (size_t)BN * 128;    // BN*128
  float* wq     = nxt + (size_t)BN * 128;   // 512
  float* wk     = wq + 512;                 // 512
  float* cq     = wk + 512;                 // B*H*N = 131072
  float* ck     = cq + (size_t)BB * NH * NN;
  float* node_w = ck + (size_t)BB * NH * NN;  // BN
  float* dualv  = node_w + BN;                // BN
  float* nw     = dualv + BN;                 // BN*16
  float* fA     = nw + (size_t)BN * DD;       // BN*16
  float* fB     = fA + (size_t)BN * DD;       // BN*16
  float* acc    = fB + (size_t)BN * DD;       // 12

  k_zero_acc<<<1, 32, 0, stream>>>(acc);
  k_encoder<<<BN / (4 * 16), 128, 0, stream>>>(emb, feat, encW1, encb1, encW2,
                                               encb2, hS);
  k_wqk<<<2, 256, 0, stream>>>(attnW, aq, ak, wq, wk);
  for (int layer = 0; layer < 2; ++layer) {
    k_cqck<<<(BB * NH * NN) / 256, 256, 0, stream>>>(hS, wq, wk, cq, ck);
    k_attn<<<BN / (2 * 16), 64, 0, stream>>>(hS, adj, cq, ck, attnW, nxt);
    k_gru<<<BN / (2 * 16), 64, 0, stream>>>(nxt, hS, gruW, gruU, grub);
  }
  k_headmlp<<<BN / (4 * 16), 128, 0, stream>>>(hS, decW1, decb1, decW2, decb2,
                                               node_w);
  k_headmlp<<<BN / (4 * 16), 128, 0, stream>>>(hS, dW1, db1, dW2, db2, dualv);
  k_pred_softmax<<<BN / 256, 256, 0, stream>>>(node_w, adj, dem, nw, fA);
  float* cur = fA;
  float* oth = fB;
  for (int it = 0; it < FLOW_ITERS; ++it) {
    k_mcf<<<BN / 256, 256, 0, stream>>>(cur, inidx, nw, dem, oth);
    float* t = cur; cur = oth; oth = t;
  }
  k_flowcost<<<(BN * DD) / 256, 256, 0, stream>>>(cur, outix, acc);
  k_dual<<<(BN * DD) / 256, 256, 0, stream>>>(dualv, adj, acc);
  k_dualdemand<<<BN / 256, 256, 0, stream>>>(dualv, dem, acc);
  k_final<<<1, 32, 0, stream>>>(acc, (float*)d_out);
}
